// LSTM2_24507083391711
// MI455X (gfx1250) — matrix operand order repaired and checked by reading, not compiled, not run on hardware
//
#include <hip/hip_runtime.h>
#include <hip/hip_bf16.h>

// ---------------------------------------------------------------------------
// 2-layer LSTM + FC + softmax for MI455X (gfx1250), bf16 WMMA everywhere.
// B=32, S=512, I=256, H=1024, O=256.
// ---------------------------------------------------------------------------

typedef __bf16 bf16;
typedef __attribute__((ext_vector_type(16))) __bf16 v16bf;
typedef __attribute__((ext_vector_type(8)))  __bf16 v8bf;
typedef __attribute__((ext_vector_type(8)))  float  v8f;

#define B_   32
#define S_   512
#define I_   256
#define H_   1024
#define O_   256
#define G4H  4096   // 4*H

// ---------------------------------------------------------------------------
// WMMA fragment loaders (wave32, 16x16x32 bf16).
//
// A fragment (16x32, M x K), A row-major with leading dim lda:
//   lane l: m = m0 + (l&15); khalf = (l>>4)*16
//   elems 0..7  = A[m][k0+khalf+0 .. +7]
//   elems 8..15 = A[m][k0+khalf+8 .. +15]
// -> two contiguous 16B loads per lane.
//
// B fragment (32x16, K x N) where B = W^T, W row-major [N][K], ld = ldk:
//   lane l: n = n0 + (l&15); kb = k0 + (l>>4)*16
//   elems e = W[n][kb+e]  -> one contiguous 32B load per lane.
// ---------------------------------------------------------------------------
__device__ __forceinline__ v16bf load_a_frag(const bf16* __restrict__ A, size_t lda,
                                             int m0, int k0, int lane) {
    int m    = m0 + (lane & 15);
    int koff = (lane >> 4) * 16;
    const bf16* p = A + (size_t)m * lda + (size_t)(k0 + koff);
    v8bf lo = *(const v8bf*)(p);
    v8bf hi = *(const v8bf*)(p + 8);
    v16bf a;
#pragma unroll
    for (int i = 0; i < 8; ++i) { a[i] = lo[i]; a[i + 8] = hi[i]; }
    return a;
}

__device__ __forceinline__ v16bf load_b_frag(const bf16* __restrict__ W, size_t ldk,
                                             int n0, int k0, int lane) {
    int n  = n0 + (lane & 15);
    int kb = k0 + (lane >> 4) * 16;
    return *(const v16bf*)(W + (size_t)n * ldk + (size_t)kb);
}

__device__ __forceinline__ v8f wmma_bf16(v16bf a, v16bf b, v8f c) {
    return __builtin_amdgcn_wmma_f32_16x16x32_bf16(false, a, false, b, (short)0, c,
                                                   false, false);
}

__device__ __forceinline__ float sigmoidf_(float x) {
    return 1.0f / (1.0f + __expf(-x));
}

// ---------------------------------------------------------------------------
// fp32 -> bf16 conversion (grid-stride)
// ---------------------------------------------------------------------------
__global__ __launch_bounds__(256) void cvt_f32_bf16(const float* __restrict__ src,
                                                    bf16* __restrict__ dst, int n) {
    int i      = blockIdx.x * blockDim.x + threadIdx.x;
    int stride = gridDim.x * blockDim.x;
    for (; i < n; i += stride) dst[i] = (bf16)src[i];
}

// ---------------------------------------------------------------------------
// out[M x N] (fp32) = A[M x K]_bf16 @ W[N x K]_bf16^T + bias1[n] + bias2[n]
// Block: 256 threads = 8 waves. Block tile 128(M) x 64(N); wave tile 16 x 64.
// grid = (N/64, M/128)
// ---------------------------------------------------------------------------
__global__ __launch_bounds__(256)
void gemm_bias_kernel(const bf16* __restrict__ A, const bf16* __restrict__ W,
                      const float* __restrict__ bias1, const float* __restrict__ bias2,
                      float* __restrict__ out, int K, int N) {
    const int lane = threadIdx.x & 31;
    const int wave = threadIdx.x >> 5;
    const int m0   = blockIdx.y * 128 + wave * 16;
    const int n0   = blockIdx.x * 64;

    v8f acc0 = {}, acc1 = {}, acc2 = {}, acc3 = {};
    for (int k0 = 0; k0 < K; k0 += 32) {
        v16bf a  = load_a_frag(A, (size_t)K, m0, k0, lane);
        v16bf b0 = load_b_frag(W, (size_t)K, n0,      k0, lane);
        v16bf b1 = load_b_frag(W, (size_t)K, n0 + 16, k0, lane);
        v16bf b2 = load_b_frag(W, (size_t)K, n0 + 32, k0, lane);
        v16bf b3 = load_b_frag(W, (size_t)K, n0 + 48, k0, lane);
        acc0 = wmma_bf16(a, b0, acc0);
        acc1 = wmma_bf16(a, b1, acc1);
        acc2 = wmma_bf16(a, b2, acc2);
        acc3 = wmma_bf16(a, b3, acc3);
    }

    const int lh = lane >> 4;
    const int nl = lane & 15;
    const int c0 = n0 + nl;
    const float bv0 = bias1[c0]      + bias2[c0];
    const float bv1 = bias1[c0 + 16] + bias2[c0 + 16];
    const float bv2 = bias1[c0 + 32] + bias2[c0 + 32];
    const float bv3 = bias1[c0 + 48] + bias2[c0 + 48];
#pragma unroll
    for (int r = 0; r < 8; ++r) {
        const int row   = m0 + r + 8 * lh;
        const size_t rb = (size_t)row * (size_t)N;
        out[rb + c0]      = acc0[r] + bv0;
        out[rb + c0 + 16] = acc1[r] + bv1;
        out[rb + c0 + 32] = acc2[r] + bv2;
        out[rb + c0 + 48] = acc3[r] + bv3;
    }
}

// ---------------------------------------------------------------------------
// One LSTM timestep (fused GEMM + gates + cell update).
//   g = xg[:, t, :] + h(t-1) @ W_hh^T ; i,f,g,o gates; c/h update.
// Block owns 16 hidden units (all 4 gate columns for them).
// 512 threads = 16 waves: (mi: 2 M-tiles) x (gate: 4) x (kh: split-K 2).
// grid = H/16 = 64 blocks.
// h(t-1) is read from out_bf at time t-1 (bf16); t==0 -> h=0, skip GEMM.
// Running c (fp32) and h (fp32) live in their final d_out slots.
// ---------------------------------------------------------------------------
__global__ __launch_bounds__(512)
void lstm_step_kernel(const float* __restrict__ xg, const bf16* __restrict__ Whh,
                      bf16* __restrict__ out_bf, float* __restrict__ h_out,
                      float* __restrict__ c_out, int t) {
    __shared__ float lds_g[4][32][16];

    const int lane = threadIdx.x & 31;
    const int wave = threadIdx.x >> 5;   // 0..15
    const int mi   = wave & 1;           // M tile (batches 0-15 / 16-31)
    const int gate = (wave >> 1) & 3;    // i,f,g,o
    const int kh   = wave >> 3;          // split-K half
    const int j0   = blockIdx.x * 16;    // hidden-unit base

    v8f acc = {};
    if (t > 0) {
        const bf16* Abase = out_bf + (size_t)(t - 1) * H_;  // row b at b*S_*H_
        const int nbase   = gate * H_ + j0;
        const int kbeg    = kh * (H_ / 2);
#pragma unroll 4
        for (int k0 = kbeg; k0 < kbeg + H_ / 2; k0 += 32) {
            v16bf a = load_a_frag(Abase, (size_t)S_ * H_, 16 * mi, k0, lane);
            v16bf b = load_b_frag(Whh, (size_t)H_, nbase, k0, lane);
            acc = wmma_bf16(a, b, acc);
        }
    }

    const int lh = lane >> 4;
    const int nl = lane & 15;
    if (kh == 0) {
#pragma unroll
        for (int r = 0; r < 8; ++r) {
            const int b = 16 * mi + r + 8 * lh;
            lds_g[gate][b][nl] = acc[r];
        }
    }
    __syncthreads();
    if (kh == 1) {
#pragma unroll
        for (int r = 0; r < 8; ++r) {
            const int b = 16 * mi + r + 8 * lh;
            lds_g[gate][b][nl] += acc[r];
        }
    }
    __syncthreads();

    // Elementwise cell update: 512 threads <-> 32 batches x 16 units.
    const int p = threadIdx.x;
    const int b = p >> 4;
    const int j = p & 15;
    const size_t xrow = ((size_t)b * S_ + (size_t)t) * (size_t)G4H + (size_t)(j0 + j);
    const float gi = lds_g[0][b][j] + xg[xrow + 0 * H_];
    const float gf = lds_g[1][b][j] + xg[xrow + 1 * H_];
    const float gg = lds_g[2][b][j] + xg[xrow + 2 * H_];
    const float go = lds_g[3][b][j] + xg[xrow + 3 * H_];

    const int hidx = b * H_ + j0 + j;
    const float cp = (t == 0) ? 0.0f : c_out[hidx];
    const float c  = sigmoidf_(gf) * cp + sigmoidf_(gi) * tanhf(gg);
    const float h  = sigmoidf_(go) * tanhf(c);

    c_out[hidx] = c;
    h_out[hidx] = h;
    out_bf[((size_t)b * S_ + (size_t)t) * (size_t)H_ + (size_t)(j0 + j)] = (bf16)h;
}

// ---------------------------------------------------------------------------
// logits[16 x 256] = A[16 x 1024]_bf16 @ Wfc[256 x 1024]^T + b_fc, then
// row softmax, written to P. Block: 256 threads = 8 waves (2 N-tiles each).
// grid = (B*S)/16 = 1024 blocks.
// ---------------------------------------------------------------------------
__global__ __launch_bounds__(256)
void fc_softmax_kernel(const bf16* __restrict__ A, const bf16* __restrict__ Wfc,
                       const float* __restrict__ bfc, float* __restrict__ P) {
    __shared__ float lds[16][O_];
    __shared__ float red[16][16];

    const int lane = threadIdx.x & 31;
    const int wave = threadIdx.x >> 5;
    const int m0   = blockIdx.x * 16;
    const int n0   = wave * 32;

    v8f acc0 = {}, acc1 = {};
    for (int k0 = 0; k0 < H_; k0 += 32) {
        v16bf a  = load_a_frag(A, (size_t)H_, m0, k0, lane);
        v16bf b0 = load_b_frag(Wfc, (size_t)H_, n0,      k0, lane);
        v16bf b1 = load_b_frag(Wfc, (size_t)H_, n0 + 16, k0, lane);
        acc0 = wmma_bf16(a, b0, acc0);
        acc1 = wmma_bf16(a, b1, acc1);
    }

    const int lh = lane >> 4;
    const int nl = lane & 15;
#pragma unroll
    for (int r = 0; r < 8; ++r) {
        const int ml = r + 8 * lh;
        lds[ml][n0 + nl]      = acc0[r] + bfc[n0 + nl];
        lds[ml][n0 + 16 + nl] = acc1[r] + bfc[n0 + 16 + nl];
    }
    __syncthreads();

    // softmax over 256 cols; 16 threads cooperate per row, 16 cols each.
    const int r   = threadIdx.x >> 4;  // row 0..15
    const int sub = threadIdx.x & 15;  // sub-range
    const int cb  = sub * 16;

    float mx = -3.402823466e+38f;
#pragma unroll
    for (int i = 0; i < 16; ++i) mx = fmaxf(mx, lds[r][cb + i]);
    red[r][sub] = mx;
    __syncthreads();
    if (sub == 0) {
        float m = red[r][0];
#pragma unroll
        for (int i = 1; i < 16; ++i) m = fmaxf(m, red[r][i]);
        red[r][0] = m;
    }
    __syncthreads();
    const float rowmax = red[r][0];
    float s = 0.0f;
#pragma unroll
    for (int i = 0; i < 16; ++i) {
        const float e = __expf(lds[r][cb + i] - rowmax);
        lds[r][cb + i] = e;
        s += e;
    }
    __syncthreads();
    red[r][sub] = s;
    __syncthreads();
    if (sub == 0) {
        float t = 0.0f;
#pragma unroll
        for (int i = 0; i < 16; ++i) t += red[r][i];
        red[r][0] = t;
    }
    __syncthreads();
    const float inv = 1.0f / red[r][0];
    float* prow = P + (size_t)(m0 + r) * (size_t)O_;
#pragma unroll
    for (int i = 0; i < 16; ++i) prow[cb + i] = lds[r][cb + i] * inv;
}

// ---------------------------------------------------------------------------
// Host-side orchestration (graph-capture safe: launches only).
// ---------------------------------------------------------------------------
extern "C" void kernel_launch(void* const* d_in, const int* in_sizes, int n_in,
                              void* d_out, int out_size, void* d_ws, size_t ws_size,
                              hipStream_t stream) {
    const float* x    = (const float*)d_in[0];
    const float* Wih1 = (const float*)d_in[1];
    const float* Whh1 = (const float*)d_in[2];
    const float* bih1 = (const float*)d_in[3];
    const float* bhh1 = (const float*)d_in[4];
    const float* Wih2 = (const float*)d_in[5];
    const float* Whh2 = (const float*)d_in[6];
    const float* bih2 = (const float*)d_in[7];
    const float* bhh2 = (const float*)d_in[8];
    const float* Wfc  = (const float*)d_in[9];
    const float* bfc  = (const float*)d_in[10];

    // ---- workspace carve-up (bf16 operands + shared fp32 xg buffer) ----
    char*  ws  = (char*)d_ws;
    size_t off = 0;
    auto alloc = [&](size_t bytes) -> void* {
        void* p = ws + off;
        off += (bytes + 255) & ~(size_t)255;
        return p;
    };
    const size_t MROWS = (size_t)B_ * S_;  // 16384
    bf16*  x_bf   = (bf16*)alloc(MROWS * I_ * 2);         //  8 MiB
    bf16*  Wih1_b = (bf16*)alloc((size_t)G4H * I_ * 2);   //  2 MiB
    bf16*  Whh1_b = (bf16*)alloc((size_t)G4H * H_ * 2);   //  8 MiB
    bf16*  Wih2_b = (bf16*)alloc((size_t)G4H * H_ * 2);   //  8 MiB
    bf16*  Whh2_b = (bf16*)alloc((size_t)G4H * H_ * 2);   //  8 MiB
    bf16*  Wfc_b  = (bf16*)alloc((size_t)O_ * H_ * 2);    //  0.5 MiB
    bf16*  out1_b = (bf16*)alloc(MROWS * H_ * 2);         // 32 MiB
    bf16*  out2_b = (bf16*)alloc(MROWS * H_ * 2);         // 32 MiB
    float* xg     = (float*)alloc(MROWS * G4H * 4);       // 256 MiB (reused L1->L2)

    // ---- d_out layout: P, h1, c1, h2, c2 (fp32, return order) ----
    float* P  = (float*)d_out;
    float* h1 = P  + (size_t)B_ * S_ * O_;   // 4,194,304
    float* c1 = h1 + (size_t)B_ * H_;
    float* h2 = c1 + (size_t)B_ * H_;
    float* c2 = h2 + (size_t)B_ * H_;

    auto cvt = [&](const float* s, bf16* d, int n) {
        int blocks = (n + 255) / 256;
        if (blocks > 4096) blocks = 4096;
        cvt_f32_bf16<<<blocks, 256, 0, stream>>>(s, d, n);
    };
    cvt(x,    x_bf,   B_ * S_ * I_);
    cvt(Wih1, Wih1_b, G4H * I_);
    cvt(Whh1, Whh1_b, G4H * H_);
    cvt(Wih2, Wih2_b, G4H * H_);
    cvt(Whh2, Whh2_b, G4H * H_);
    cvt(Wfc,  Wfc_b,  O_ * H_);

    // ---- layer 1 ----
    gemm_bias_kernel<<<dim3(G4H / 64, (int)(MROWS / 128)), 256, 0, stream>>>(
        x_bf, Wih1_b, bih1, bhh1, xg, I_, G4H);
    for (int t = 0; t < S_; ++t)
        lstm_step_kernel<<<H_ / 16, 512, 0, stream>>>(xg, Whh1_b, out1_b, h1, c1, t);

    // ---- layer 2 (xg buffer reused) ----
    gemm_bias_kernel<<<dim3(G4H / 64, (int)(MROWS / 128)), 256, 0, stream>>>(
        out1_b, Wih2_b, bih2, bhh2, xg, H_, G4H);
    for (int t = 0; t < S_; ++t)
        lstm_step_kernel<<<H_ / 16, 512, 0, stream>>>(xg, Whh2_b, out2_b, h2, c2, t);

    // ---- FC + softmax ----
    fc_softmax_kernel<<<(int)(MROWS / 16), 256, 0, stream>>>(out2_b, Wfc_b, bfc, P);
}
